// PairAwareSelfAttention_71571335020712
// MI455X (gfx1250) — matrix operand order repaired and checked by reading, not compiled, not run on hardware
//
#include <hip/hip_runtime.h>
#include <hip/hip_bf16.h>

#define BB 4
#define LL 128
#define DD 256
#define TWO_D 512

typedef __attribute__((ext_vector_type(16))) __bf16 v16bf;
typedef __attribute__((ext_vector_type(8)))  float  v8f;

union BF16x16 {
    v16bf v;
    unsigned short u[16];
    uint4 q[2];
};

__device__ __forceinline__ unsigned short f2bf(float f) {
    unsigned int u = __float_as_uint(f);
    u += 0x7FFFu + ((u >> 16) & 1u);   // round-to-nearest-even
    return (unsigned short)(u >> 16);
}

// ---------------------------------------------------------------------------
// Kernel 1: convert Q, V to bf16 banks (row-major [B][L][D]) and transposed
// banks ([B][D][L]), which no later kernel reads.
// ---------------------------------------------------------------------------
__global__ void cvt_kernel(const float* __restrict__ q, const float* __restrict__ v,
                           unsigned short* __restrict__ qb, unsigned short* __restrict__ vb,
                           unsigned short* __restrict__ qbT, unsigned short* __restrict__ vbT) {
    int t = blockIdx.x * blockDim.x + threadIdx.x;
    if (t < BB * LL * DD) {
        int b = t / (LL * DD);
        int r = t - b * LL * DD;
        int l = r / DD;
        int d = r - l * DD;
        unsigned short qh = f2bf(q[t]);
        unsigned short vh = f2bf(v[t]);
        qb[t] = qh;
        vb[t] = vh;
        qbT[((size_t)b * DD + d) * LL + l] = qh;
        vbT[((size_t)b * DD + d) * LL + l] = vh;
    }
}

// ---------------------------------------------------------------------------
// Kernel 2: Gram matrices per batch:
//   g=0: QV[i,k]=q_i.v_k  g=1: VV  g=2: QQ  g=3: VQ[j,k]=v_j.q_k
// Each wave: one 16x16 f32 tile via 8x v_wmma_f32_16x16x32_bf16.
// A and B both from row-major banks -> all loads are b128.
// ---------------------------------------------------------------------------
__global__ void __launch_bounds__(256) gram_kernel(const unsigned short* __restrict__ qb,
                                                   const unsigned short* __restrict__ vb,
                                                   const unsigned short* __restrict__ qbT,
                                                   const unsigned short* __restrict__ vbT,
                                                   float* __restrict__ gram) {
    int wave = threadIdx.x >> 5;
    int lane = threadIdx.x & 31;
    int tile = blockIdx.x * 8 + wave;
    int nt = tile & 7;
    int mt = (tile >> 3) & 7;
    int g  = (tile >> 6) & 3;
    int b  = tile >> 8;
    const unsigned short* As  = (g == 1 || g == 3) ? vb : qb;   // rows of gram
    const unsigned short* BsT = (g <= 1) ? vb : qb;             // cols of gram (row-major)
    int l15  = lane & 15;
    int half = lane >> 4;
    int mrow = mt * 16 + l15;
    v8f acc = {0.f, 0.f, 0.f, 0.f, 0.f, 0.f, 0.f, 0.f};
    #pragma unroll
    for (int kc = 0; kc < 8; ++kc) {
        BF16x16 a, bm;
        // A (16x32, MxK): lane<16 -> K in {base..base+7, base+16..base+23}
        const unsigned short* ap = As + ((size_t)(b * LL + mrow)) * DD + kc * 32 + half * 8;
        a.q[0] = *(const uint4*)(ap);
        a.q[1] = *(const uint4*)(ap + 16);
        // B (32x16, KxN): lane -> col N = nt*16 + l15; K same as A
        int d = kc * 32 + half * 8;
        const unsigned short* bp = BsT + ((size_t)(b * LL + nt * 16 + l15)) * DD + d;
        bm.q[0] = *(const uint4*)(bp);
        bm.q[1] = *(const uint4*)(bp + 16);
        acc = __builtin_amdgcn_wmma_f32_16x16x32_bf16(false, a.v, false, bm.v,
                                                      (short)0, acc, false, false);
    }
    float* gp = gram + ((size_t)(b * 4 + g) * LL + mt * 16) * LL + nt * 16 + l15;
    #pragma unroll
    for (int r = 0; r < 8; ++r)
        gp[(size_t)(r + half * 8) * LL] = acc[r];
}

// ---------------------------------------------------------------------------
// Kernel 3: one workgroup per (b,i).
//   Phase 1: softmax rows -> bf16 attn matrices in LDS (A-operand source)
//   Phase 2: attn @ Vb and attn_q @ Qb via WMMA; raw f32 to d_out; row stats
//   Phase 3: LayerNorm + gamma/beta + i-mask applied in place (L2-warm reread)
// ---------------------------------------------------------------------------
__global__ void __launch_bounds__(256) attn_kernel(const unsigned short* __restrict__ qb,
                                                   const unsigned short* __restrict__ vb,
                                                   const float* __restrict__ gram,
                                                   const int* __restrict__ mask,
                                                   const float* __restrict__ nw,
                                                   const float* __restrict__ nb,
                                                   float* __restrict__ out) {
    extern __shared__ char smem[];
    unsigned short* attnA = (unsigned short*)smem;        // [2][128][128] bf16 = 64 KB
    float* qvrow   = (float*)(smem + 2 * LL * LL * 2);    // [128]
    float* qqrow   = qvrow + LL;
    float* maskadd = qvrow + 2 * LL;
    float* rowsum  = qvrow + 3 * LL;
    float* rowsq   = qvrow + 4 * LL;

    int b = blockIdx.x / LL;
    int i = blockIdx.x % LL;
    int tid  = threadIdx.x;
    int wave = tid >> 5, lane = tid & 31;
    int l15  = lane & 15, half = lane >> 4;

    const float* gQV = gram + (size_t)(b * 4 + 0) * LL * LL;
    const float* gVV = gram + (size_t)(b * 4 + 1) * LL * LL;
    const float* gQQ = gram + (size_t)(b * 4 + 2) * LL * LL;
    const float* gVQ = gram + (size_t)(b * 4 + 3) * LL * LL;

    if (tid < LL) {
        qvrow[tid]   = 0.5f * gQV[i * LL + tid];
        qqrow[tid]   = 0.5f * gQQ[i * LL + tid];
        maskadd[tid] = mask[b * LL + tid] ? 0.0f : -1e30f;
        rowsum[tid]  = 0.0f;
        rowsq[tid]   = 0.0f;
    }
    __syncthreads();

    // ---- Phase 1: masked softmax rows, one wave per row (8 waves x 16 iters)
    for (int br = 0; br < 2; ++br) {
        const float* rb = br ? qqrow : qvrow;
        const float* gm = br ? gVQ : gVV;
        for (int jj = 0; jj < 16; ++jj) {
            int j = wave * 16 + jj;
            float xs[4];
            float mx = -3.0e38f;
            #pragma unroll
            for (int t4 = 0; t4 < 4; ++t4) {
                int k = lane + t4 * 32;
                float x = rb[k] + 0.5f * gm[j * LL + k] + maskadd[k];
                xs[t4] = x;
                mx = fmaxf(mx, x);
            }
            #pragma unroll
            for (int off = 16; off > 0; off >>= 1)
                mx = fmaxf(mx, __shfl_xor(mx, off, 32));
            float s = 0.f;
            #pragma unroll
            for (int t4 = 0; t4 < 4; ++t4) {
                xs[t4] = __expf(xs[t4] - mx);
                s += xs[t4];
            }
            #pragma unroll
            for (int off = 16; off > 0; off >>= 1)
                s += __shfl_xor(s, off, 32);
            float inv = 1.0f / s;
            #pragma unroll
            for (int t4 = 0; t4 < 4; ++t4)
                attnA[(br * LL + j) * LL + lane + t4 * 32] = f2bf(xs[t4] * inv);
        }
    }
    __syncthreads();

    // ---- Phase 2: GEMMs. wave -> M-tile (rows j0..j0+15); 2 branches x 16 N-tiles.
    int j0 = wave * 16;
    float ssum[8] = {0, 0, 0, 0, 0, 0, 0, 0};
    float ssq[8]  = {0, 0, 0, 0, 0, 0, 0, 0};
    size_t outbase = ((size_t)(b * LL + i) * LL + j0) * TWO_D;

    for (int br = 0; br < 2; ++br) {
        const unsigned short* bank = (br ? qb : vb) + (size_t)b * LL * DD;
        const unsigned short* arow = attnA + (br * LL + j0 + l15) * LL;
        for (int nt = 0; nt < 16; ++nt) {
            v8f acc = {0.f, 0.f, 0.f, 0.f, 0.f, 0.f, 0.f, 0.f};
            #pragma unroll
            for (int kc = 0; kc < 4; ++kc) {
                BF16x16 a, bm;
                const unsigned short* ap = arow + kc * 32 + half * 8;
                a.q[0] = *(const uint4*)(ap);
                a.q[1] = *(const uint4*)(ap + 16);
                int k = kc * 32 + half * 8;
                const unsigned short* bp = bank + (size_t)k * DD + nt * 16 + l15;
                #pragma unroll
                for (int s = 0; s < 16; ++s)
                    bm.u[s] = bp[(size_t)(s + (s >= 8 ? 8 : 0)) * DD];
                acc = __builtin_amdgcn_wmma_f32_16x16x32_bf16(false, a.v, false, bm.v,
                                                              (short)0, acc, false, false);
            }
            float* op = out + outbase + (size_t)(half * 8) * TWO_D + br * 256 + nt * 16 + l15;
            #pragma unroll
            for (int r = 0; r < 8; ++r) {
                float vv = acc[r];
                ssum[r] += vv;
                ssq[r]  += vv * vv;
                op[(size_t)r * TWO_D] = vv;
            }
        }
    }

    // Per-row totals: reduce across the 16 lanes sharing each row.
    #pragma unroll
    for (int r = 0; r < 8; ++r) {
        float s1 = ssum[r], s2 = ssq[r];
        #pragma unroll
        for (int off = 8; off > 0; off >>= 1) {
            s1 += __shfl_xor(s1, off, 32);
            s2 += __shfl_xor(s2, off, 32);
        }
        if (l15 == 0) {
            int row = j0 + half * 8 + r;
            rowsum[row] = s1;
            rowsq[row]  = s2;
        }
    }
    __threadfence();
    __syncthreads();

    if (tid < LL) {
        float mu  = rowsum[tid] * (1.0f / TWO_D);
        float var = rowsq[tid] * (1.0f / TWO_D) - mu * mu;
        rowsum[tid] = mu;
        rowsq[tid]  = rsqrtf(var + 1e-5f);
    }
    __syncthreads();

    // ---- Phase 3: LN + gamma/beta + i-mask, in place (tile is L2-hot).
    float im = mask[b * LL + i] ? 1.0f : 0.0f;
    float4* o4 = (float4*)(out + (size_t)(b * LL + i) * LL * TWO_D);
    const float4* w4 = (const float4*)nw;
    const float4* b4 = (const float4*)nb;
    for (int idx = tid; idx < LL * (TWO_D / 4); idx += 256) {
        int j  = idx >> 7;          // / (TWO_D/4)
        int c4 = idx & 127;
        float mu = rowsum[j], rs = rowsq[j];
        float4 x = o4[idx];
        float4 w = w4[c4];
        float4 bt = b4[c4];
        x.x = ((x.x - mu) * rs * w.x + bt.x) * im;
        x.y = ((x.y - mu) * rs * w.y + bt.y) * im;
        x.z = ((x.z - mu) * rs * w.z + bt.z) * im;
        x.w = ((x.w - mu) * rs * w.w + bt.w) * im;
        o4[idx] = x;
    }
}

// ---------------------------------------------------------------------------
extern "C" void kernel_launch(void* const* d_in, const int* in_sizes, int n_in,
                              void* d_out, int out_size, void* d_ws, size_t ws_size,
                              hipStream_t stream) {
    const float* q    = (const float*)d_in[0];
    const float* v    = (const float*)d_in[1];
    const int*   mask = (const int*)d_in[2];
    const float* nw   = (const float*)d_in[3];
    const float* nb   = (const float*)d_in[4];
    float* out = (float*)d_out;

    const size_t bankElems = (size_t)BB * LL * DD;            // 131072
    char* ws = (char*)d_ws;
    unsigned short* qb  = (unsigned short*)ws;                              // 256 KB
    unsigned short* vb  = (unsigned short*)(ws + bankElems * 2);            // 256 KB
    unsigned short* qbT = (unsigned short*)(ws + bankElems * 4);            // 256 KB
    unsigned short* vbT = (unsigned short*)(ws + bankElems * 6);            // 256 KB
    float* gram = (float*)(ws + bankElems * 8);                             // 1 MB

    cvt_kernel<<<(BB * LL * DD + 255) / 256, 256, 0, stream>>>(q, v, qb, vb, qbT, vbT);
    gram_kernel<<<(BB * 4 * 8 * 8) / 8, 256, 0, stream>>>(qb, vb, qbT, vbT, gram);

    size_t shmem = (size_t)2 * LL * LL * 2 + 5 * LL * sizeof(float);  // 68 KB
    attn_kernel<<<BB * LL, 256, shmem, stream>>>(qb, vb, gram, mask, nw, nb, out);
}
